// SFM_Net_19121194401841
// MI455X (gfx1250) — matrix operand order repaired and checked by reading, not compiled, not run on hardware
//
#include <hip/hip_runtime.h>
#include <hip/hip_bf16.h>
#include <math.h>
#include <stdint.h>

typedef __attribute__((ext_vector_type(16))) __bf16 v16bf;
typedef __attribute__((ext_vector_type(8)))  float  v8f;

#define HID   64
#define FREQ  10
#define OUTD  16
#define BSZ   1024
#define TSTEP 128
#define INDIM 256
#define NCAT  266      // 64(i) + 64(ste) + 64(c) + 64(o) + 10(fre)
#define NPAD  272      // padded to 17 tiles of 16
#define NTILE 17
#define TWO_PI_F 6.2831853071795864769f

// Packed-fragment geometry: per (k-block, n-tile) tile of 32x16, lane l's 16
// bf16 B-fragment elements are contiguous: elem e of lane l is
// (K = kb + (l>>4)*16 + e, N = nt*16 + (l&15)).
#define TILE_ELEMS   512                         // 32 lanes * 16 elems
#define PACKW_TILES  (8 * NTILE)                 // K=256 -> 8 k-blocks
#define PACKU_TILES  (2 * NTILE)                 // K=64  -> 2 k-blocks
#define PACKW_ELEMS  (PACKW_TILES * TILE_ELEMS)  // 69632
#define PACKU_ELEMS  (PACKU_TILES * TILE_ELEMS)  // 17408

__device__ __forceinline__ float hsig(float x) {
    return fminf(fmaxf(x * (1.0f / 6.0f) + 0.5f, 0.0f), 1.0f);
}

__device__ __forceinline__ float fetchW(const float* Wi, const float* Wste, const float* Wc,
                                        const float* Wo, const float* Wfre, int k, int n) {
    if (n < 64)   return Wi [k * 64 + n];
    if (n < 128)  return Wste[k * 64 + (n - 64)];
    if (n < 192)  return Wc [k * 64 + (n - 128)];
    if (n < 256)  return Wo [k * 64 + (n - 192)];
    if (n < NCAT) return Wfre[k * FREQ + (n - 256)];
    return 0.0f;
}

__device__ __forceinline__ float fetchU(const float* Ui, const float* Uste, const float* Uc,
                                        const float* Uo, const float* Ufre, int k, int n) {
    if (n < 64)   return Ui  [k * 64 + n];
    if (n < 128)  return Uste[k * 64 + (n - 64)];
    if (n < 192)  return Uc  [k * 64 + (n - 128)];
    if (n < 256)  return Uo  [k * 64 + (n - 192)];
    if (n < NCAT) return Ufre[k * FREQ + (n - 256)];
    return 0.0f;
}

// ---------------------------------------------------------------------------
// Kernel 0 (one-shot, tiny): pack weights into WMMA B-fragment lane order +
// flat bias vector. All branchy gathers live here, outside the hot loops.
// ---------------------------------------------------------------------------
__global__ void __launch_bounds__(256)
sfm_pack(const float* __restrict__ Wi, const float* __restrict__ Wste,
         const float* __restrict__ Wfre, const float* __restrict__ Wc,
         const float* __restrict__ Wo,
         const float* __restrict__ Ui, const float* __restrict__ Uste,
         const float* __restrict__ Ufre, const float* __restrict__ Uc,
         const float* __restrict__ Uo,
         const float* __restrict__ bi, const float* __restrict__ bste,
         const float* __restrict__ bfre, const float* __restrict__ bc,
         const float* __restrict__ bo,
         __bf16* __restrict__ packW, __bf16* __restrict__ packU,
         float* __restrict__ biasP) {
    const int idx = blockIdx.x * 256 + threadIdx.x;
    if (idx < PACKW_ELEMS) {
        const int tile = idx >> 9, e = idx & 15, l = (idx >> 4) & 31;
        const int kbi = tile / NTILE, nt = tile % NTILE;
        const int K = kbi * 32 + (l >> 4) * 16 + e;
        const int N = nt * 16 + (l & 15);
        packW[idx] = (__bf16)fetchW(Wi, Wste, Wc, Wo, Wfre, K, N);
    } else if (idx < PACKW_ELEMS + PACKU_ELEMS) {
        const int j = idx - PACKW_ELEMS;
        const int tile = j >> 9, e = j & 15, l = (j >> 4) & 31;
        const int kbi = tile / NTILE, nt = tile % NTILE;
        const int K = kbi * 32 + (l >> 4) * 16 + e;
        const int N = nt * 16 + (l & 15);
        packU[j] = (__bf16)fetchU(Ui, Uste, Uc, Uo, Ufre, K, N);
    } else if (idx < PACKW_ELEMS + PACKU_ELEMS + NPAD) {
        const int n = idx - PACKW_ELEMS - PACKU_ELEMS;
        float v;
        if      (n < 64)   v = bi [n];
        else if (n < 128)  v = bste[n - 64];
        else if (n < 192)  v = bc [n - 128];
        else if (n < 256)  v = bo [n - 192];
        else if (n < NCAT) v = bfre[n - 256];
        else               v = 0.0f;
        biasP[n] = v;
    }
}

// ---------------------------------------------------------------------------
// Kernel A: xw = x @ Wcat + bias.  B fragments = single 32B vector loads from
// packed (L2-resident) weights; A fragments from LDS-contiguous bf16 tile.
// ---------------------------------------------------------------------------
__global__ void __launch_bounds__(128)
sfm_pregemm(const float* __restrict__ g1,
            const __bf16* __restrict__ packW,
            const float* __restrict__ biasP,
            float* __restrict__ xw) {
    __shared__ __bf16 xs[16][INDIM];          // 8 KB

    const int  tid  = threadIdx.x;
    const long row0 = (long)blockIdx.x * 16;

    // float4 loads of the x tile, convert to bf16
    const float4* g4 = (const float4*)(g1 + row0 * INDIM);
    for (int idx = tid; idx < 16 * (INDIM / 4); idx += 128) {
        const int r = idx >> 6, c4 = (idx & 63) << 2;
        const float4 v = g4[idx];
        xs[r][c4 + 0] = (__bf16)v.x;
        xs[r][c4 + 1] = (__bf16)v.y;
        xs[r][c4 + 2] = (__bf16)v.z;
        xs[r][c4 + 3] = (__bf16)v.w;
    }
    __syncthreads();

    const int lane = tid & 31;
    const int wave = tid >> 5;
    const int mrow = lane & 15;
    const int hi   = lane >> 4;

    for (int nt = wave; nt < NTILE; nt += 4) {
        const int ncol = nt * 16 + mrow;
        v8f acc = {};
        #pragma unroll
        for (int kbi = 0; kbi < 8; ++kbi) {
            const int kb = kbi << 5;
            v16bf a;
            #pragma unroll
            for (int e = 0; e < 8; ++e) {
                a[e]     = xs[mrow][kb + hi * 16 + e];
                a[8 + e] = xs[mrow][kb + hi * 16 + 8 + e];
            }
            const v16bf b = *(const v16bf*)(packW + (((kbi * NTILE) + nt) << 9) + (lane << 4));
            acc = __builtin_amdgcn_wmma_f32_16x16x32_bf16(false, a, false, b,
                                                          (short)0, acc, false, false);
        }
        const float bias = biasP[ncol];
        #pragma unroll
        for (int r = 0; r < 8; ++r) {
            xw[(row0 + r + hi * 8) * NPAD + ncol] = acc[r] + bias;
        }
    }
}

// ---------------------------------------------------------------------------
// Kernel B: recurrence. One block = 16 batch rows for all T.
// xwS staged with gfx1250 async global->LDS (ASYNCcnt); h@Ucat via WMMA.
// sin/cos come from an exact 10-entry phase table (omega is periodic mod 10),
// so the only hot-loop transcendentals are the 8 tanhf per thread-step.
// ---------------------------------------------------------------------------
__global__ void __launch_bounds__(256)
sfm_recurrent(const float* __restrict__ xw,
              const __bf16* __restrict__ packU,
              const float* __restrict__ Ua, const float* __restrict__ ba,
              float* __restrict__ hT) {
    __shared__ alignas(32) __bf16 UbfL[PACKU_ELEMS]; // 34.8 KB, fragment order
    __shared__ __bf16 hbf[16][HID];                  //  2   KB
    __shared__ float  hU [16][NPAD];                 // 17.4 KB
    __shared__ alignas(16) float xwS[16][NPAD];      // 17.4 KB
    __shared__ float  cosTab[FREQ], sinTab[FREQ];

    const int tid = threadIdx.x;        // 256 threads, 8 waves
    const int b0  = blockIdx.x * 16;

    // Contiguous copy of packed U fragments into LDS (no branches)
    const uint4* pu4 = (const uint4*)packU;
    uint4* ub4 = (uint4*)UbfL;
    for (int idx = tid; idx < (PACKU_ELEMS * 2) / 16; idx += 256)
        ub4[idx] = pu4[idx];
    for (int idx = tid; idx < 16 * HID; idx += 256)
        hbf[idx >> 6][idx & 63] = (__bf16)0.0f;
    if (tid < FREQ) {
        float s, c;
        __sincosf(TWO_PI_F * (float)tid * 0.1f, &s, &c);
        cosTab[tid] = c;
        sinTab[tid] = s;
    }

    float Sre[4][FREQ], Sim[4][FREQ], hcur[4];
    int rowp[4], hp[4];
    float bap[4];
    #pragma unroll
    for (int p = 0; p < 4; ++p) {
        const int idx = p * 256 + tid;
        rowp[p] = idx >> 6;
        hp[p]   = idx & 63;
        bap[p]  = ba[hp[p]];
        hcur[p] = 0.0f;
        #pragma unroll
        for (int f = 0; f < FREQ; ++f) { Sre[p][f] = 0.0f; Sim[p][f] = 0.0f; }
    }
    float UaR[FREQ];
    int   ph[FREQ];                      // phase index: ((t+1)*f) mod 10
    #pragma unroll
    for (int f = 0; f < FREQ; ++f) { UaR[f] = Ua[f]; ph[f] = f; }

    const int lane = tid & 31;
    const int wave = tid >> 5;
    const int mrow = lane & 15;
    const int hi   = lane >> 4;

    __syncthreads();

    for (int t = 0; t < TSTEP; ++t) {
        // ---- async global->LDS staging of this step's x-projections ----
        // 16 rows x 272 f32 = 1088 16-byte transfers
        for (int idx = tid; idx < 1088; idx += 256) {
            const int r = idx / 68, c4 = (idx % 68) << 2;
            const unsigned ldsOff = (unsigned)(uintptr_t)&xwS[r][c4];
            const float* gsrc = &xw[((long)(b0 + r) * TSTEP + t) * NPAD + c4];
            asm volatile("global_load_async_to_lds_b128 %0, %1, off"
                         :
                         : "v"(ldsOff), "v"(gsrc)
                         : "memory");
        }
        asm volatile("s_wait_asynccnt 0x0" ::: "memory");
        __syncthreads();

        // ---- hU = h @ [U_i|U_ste|U_c|U_o|U_fre]  (16x64 @ 64x272) ----
        for (int nt = wave; nt < NTILE; nt += 8) {
            const int ncol = nt * 16 + mrow;
            v8f acc = {};
            #pragma unroll
            for (int kbi = 0; kbi < 2; ++kbi) {
                const int kb = kbi << 5;
                v16bf a;
                #pragma unroll
                for (int e = 0; e < 8; ++e) {
                    a[e]     = hbf[mrow][kb + hi * 16 + e];
                    a[8 + e] = hbf[mrow][kb + hi * 16 + 8 + e];
                }
                const v16bf b = *(const v16bf*)(&UbfL[(((kbi * NTILE) + nt) << 9) + (lane << 4)]);
                acc = __builtin_amdgcn_wmma_f32_16x16x32_bf16(false, a, false, b,
                                                              (short)0, acc, false, false);
            }
            #pragma unroll
            for (int r = 0; r < 8; ++r)
                hU[r + hi * 8][ncol] = acc[r];
        }
        __syncthreads();

        // ---- per-step sin/cos from the exact phase table ----
        float cw[FREQ], sw[FREQ];
        #pragma unroll
        for (int f = 0; f < FREQ; ++f) {
            cw[f] = cosTab[ph[f]];
            sw[f] = sinTab[ph[f]];
            ph[f] += f;
            if (ph[f] >= FREQ) ph[f] -= FREQ;
        }

        // ---- gates + oscillator update (f32) ----
        #pragma unroll
        for (int p = 0; p < 4; ++p) {
            const int row = rowp[p];
            const int h   = hp[p];
            const float ig  = hsig(xwS[row][h]        + hU[row][h]);
            const float ste = hsig(xwS[row][64 + h]   + hU[row][64 + h]);
            const float cg  = ig * tanhf(xwS[row][128 + h] + hU[row][128 + h]);
            const float og  = hsig(xwS[row][192 + h]  + hU[row][192 + h]);
            float acc = 0.0f;
            #pragma unroll
            for (int f = 0; f < FREQ; ++f) {
                const float fre = hsig(xwS[row][256 + f] + hU[row][256 + f]);
                const float ff  = ste * fre;
                const float sre = ff * Sre[p][f] + cg * cw[f];
                const float sim = ff * Sim[p][f] + cg * sw[f];
                Sre[p][f] = sre;
                Sim[p][f] = sim;
                acc += (sre * sre + sim * sim) * UaR[f];
            }
            hcur[p] = og * tanhf(acc + bap[p]);
        }
        __syncthreads();
        #pragma unroll
        for (int p = 0; p < 4; ++p) {
            const int idx = p * 256 + tid;
            hbf[idx >> 6][idx & 63] = (__bf16)hcur[p];
        }
        __syncthreads();
    }

    #pragma unroll
    for (int p = 0; p < 4; ++p) {
        hT[(long)(b0 + rowp[p]) * HID + hp[p]] = hcur[p];
    }
}

// ---------------------------------------------------------------------------
// Kernel C: out[b] = (hT[b] @ W_p + b_p) @ fc_w + fc_b
// ---------------------------------------------------------------------------
__global__ void __launch_bounds__(256)
sfm_head(const float* __restrict__ hT, const float* __restrict__ Wp,
         const float* __restrict__ bp, const float* __restrict__ fcw,
         const float* __restrict__ fcb, float* __restrict__ out) {
    const int b = blockIdx.x * blockDim.x + threadIdx.x;
    if (b >= BSZ) return;
    float acc = 0.0f;
    #pragma unroll
    for (int o = 0; o < OUTD; ++o) {
        float p = bp[o];
        for (int h = 0; h < HID; ++h) p += hT[b * HID + h] * Wp[h * OUTD + o];
        acc += p * fcw[o];
    }
    out[b] = acc + fcb[0];
}

// ---------------------------------------------------------------------------
extern "C" void kernel_launch(void* const* d_in, const int* in_sizes, int n_in,
                              void* d_out, int out_size, void* d_ws, size_t ws_size,
                              hipStream_t stream) {
    const float* g1    = (const float*)d_in[0];
    const float* W_i   = (const float*)d_in[1];
    const float* U_i   = (const float*)d_in[2];
    const float* b_i   = (const float*)d_in[3];
    const float* W_ste = (const float*)d_in[4];
    const float* U_ste = (const float*)d_in[5];
    const float* b_ste = (const float*)d_in[6];
    const float* W_fre = (const float*)d_in[7];
    const float* U_fre = (const float*)d_in[8];
    const float* b_fre = (const float*)d_in[9];
    const float* W_c   = (const float*)d_in[10];
    const float* U_c   = (const float*)d_in[11];
    const float* b_c   = (const float*)d_in[12];
    const float* W_o   = (const float*)d_in[13];
    const float* U_o   = (const float*)d_in[14];
    const float* b_o   = (const float*)d_in[15];
    const float* U_a   = (const float*)d_in[16];
    const float* b_a   = (const float*)d_in[17];
    const float* W_p   = (const float*)d_in[18];
    const float* b_p   = (const float*)d_in[19];
    const float* fc_w  = (const float*)d_in[20];
    const float* fc_b  = (const float*)d_in[21];

    // Workspace carve-up (keep 16B/32B alignment for vector loads)
    float*  xw    = (float*)d_ws;                              // BS*T x NPAD f32
    float*  hT    = xw + (size_t)BSZ * TSTEP * NPAD;           // BS x HID f32
    __bf16* packW = (__bf16*)(hT + (size_t)BSZ * HID);         // 69632 bf16
    __bf16* packU = packW + PACKW_ELEMS;                       // 17408 bf16
    float*  biasP = (float*)(packU + PACKU_ELEMS);             // 272 f32
    float*  out   = (float*)d_out;

    // Phase 0: one-shot weight packing (B-fragment lane order)
    const int packTotal = PACKW_ELEMS + PACKU_ELEMS + NPAD;
    sfm_pack<<<dim3((packTotal + 255) / 256), dim3(256), 0, stream>>>(
        W_i, W_ste, W_fre, W_c, W_o,
        U_i, U_ste, U_fre, U_c, U_o,
        b_i, b_ste, b_fre, b_c, b_o,
        packW, packU, biasP);

    // Phase 1: hoisted input GEMM (memory-bound; one pass over g1)
    sfm_pregemm<<<dim3((BSZ * TSTEP) / 16), dim3(128), 0, stream>>>(
        g1, packW, biasP, xw);

    // Phase 2: recurrence, 64 independent batch-slice blocks
    sfm_recurrent<<<dim3(BSZ / 16), dim3(256), 0, stream>>>(
        xw, packU, U_a, b_a, hT);

    // Phase 3: head
    sfm_head<<<dim3((BSZ + 255) / 256), dim3(256), 0, stream>>>(
        hT, W_p, b_p, fc_w, fc_b, out);
}
